// BiMambaBlock_34179349741577
// MI455X (gfx1250) — matrix operand order repaired and checked by reading, not compiled, not run on hardware
//
#include <hip/hip_runtime.h>
#include <hip/hip_bf16.h>
#include <math.h>

typedef __attribute__((ext_vector_type(16))) _Float16 v16h;
typedef __attribute__((ext_vector_type(8)))  _Float16 v8h;
typedef __attribute__((ext_vector_type(8)))  float    v8f;

#define SEQL 4096
#define DIN  64
#define DI   128
#define NST  16
#define NCH  64   // scan chunks
#define CL   64   // chunk length (NCH*CL == SEQL)

__device__ __forceinline__ v8f wmma32(v16h a, v16h b, v8f c) {
  return __builtin_amdgcn_wmma_f32_16x16x32_f16(false, a, false, b, (short)0, c,
                                                false, false);
}
__device__ __forceinline__ v16h cat16(v8h lo, v8h hi) {
  return __builtin_shufflevector(lo, hi, 0,1,2,3,4,5,6,7,8,9,10,11,12,13,14,15);
}

// ---- x transpose to f16 [b][l][d] (one shot; feeds in_proj x2 + res) -------
__global__ __launch_bounds__(256) void k_packx(const float* __restrict__ x,
                                               _Float16* __restrict__ xh)
{
  int idx = blockIdx.x * blockDim.x + threadIdx.x;
  if (idx >= 4 * SEQL * DIN) return;
  int d  = idx & (DIN - 1);
  int bl = idx >> 6;
  int l  = bl & (SEQL - 1);
  int bb = bl >> 12;
  xh[idx] = (_Float16)x[((size_t)bb * DIN + d) * SEQL + l];
}

// ---- generic weight prepack into WMMA B-fragment order ---------------------
__global__ __launch_bounds__(256) void k_pack(const float* __restrict__ src,
    _Float16* __restrict__ dst, int NT, int KK, int K, int maxrow)
{
  int t = blockIdx.x * blockDim.x + threadIdx.x;
  if (t >= NT * KK * 32) return;
  int lane = t & 31;
  int kk   = (t >> 5) % KK;
  int nt   = t / (32 * KK);
  int am   = lane & 15;
  int kbB  = (lane & 16) ? 8 : 0;
  int n    = nt * 16 + am; if (n > maxrow) n = maxrow;
  const float* row = src + (size_t)n * K + kk * 32 + kbB;
  _Float16* o = dst + (size_t)t * 16;
#pragma unroll
  for (int i = 0; i < 16; ++i) o[i] = (_Float16)row[i + (i >= 8 ? 8 : 0)];
}

// ---- conv weight prepack: frag = ((pos*4 + kk)*4 + nt)*32 + lane -----------
__global__ __launch_bounds__(256) void k_packconv(const float* __restrict__ cw,
                                                  _Float16* __restrict__ dst)
{
  int t = blockIdx.x * blockDim.x + threadIdx.x;
  if (t >= 9 * 4 * 4 * 32) return;
  int lane = t & 31;
  int nt   = (t >> 5) & 3;
  int kk   = (t >> 7) & 3;
  int pos  = t >> 9;
  int ky = pos / 3, kx = pos % 3;
  int am  = lane & 15;
  int kbB = (lane & 16) ? 8 : 0;
  int n   = nt * 16 + am;
  _Float16* o = dst + (size_t)t * 16;
#pragma unroll
  for (int i = 0; i < 16; ++i) {
    int kb = kk * 32 + kbB + i + (i >= 8 ? 8 : 0);
    o[i] = (_Float16)cw[((n * DI + kb) * 3 + ky) * 3 + kx];
  }
}

// ---------------- in_proj: one wave = 16 M-rows x all 256 N, K=64 ----------
__global__ __launch_bounds__(32) void k_inproj(const _Float16* __restrict__ xh,
    const _Float16* __restrict__ wp, float* __restrict__ xi,
    float* __restrict__ zb, int rev)
{
  const int lane = threadIdx.x;
  const int am  = lane & 15;
  const int hi  = (lane >> 4) & 1;
  const int kbA = hi ? 8 : 0;
  const int m   = blockIdx.x * 16 + am;
  const int bb  = m >> 12;
  const int l   = m & (SEQL - 1);
  const int lp  = rev ? (SEQL - 1 - l) : l;
  const _Float16* arow = xh + ((size_t)bb * SEQL + lp) * DIN;
  const v16h* bp = (const v16h*)wp;

  v16h a0 = cat16(*(const v8h*)(arow + kbA),      *(const v8h*)(arow + 16 + kbA));
  v16h a1 = cat16(*(const v8h*)(arow + 32 + kbA), *(const v8h*)(arow + 48 + kbA));

  for (int nt = 0; nt < 16; ++nt) {
    v16h b0 = bp[(nt * 2 + 0) * 32 + lane];
    v16h b1 = bp[(nt * 2 + 1) * 32 + lane];
    v8f acc = {};
    acc = wmma32(a0, b0, acc);
    acc = wmma32(a1, b1, acc);
    float* dst = (nt < 8) ? xi : zb;           // uniform per N-tile
    int col = nt * 16 + am - ((nt < 8) ? 0 : DI);
#pragma unroll
    for (int r = 0; r < 8; ++r) {
      int mm = blockIdx.x * 16 + r + (hi ? 8 : 0);
      int bb2 = mm >> 12, l2 = mm & (SEQL - 1);
      dst[((size_t)bb2 * SEQL + l2) * DI + col] = acc[r];
    }
  }
}

// ---------------- residual 1x1: N=64, K=64 ---------------------------------
__global__ __launch_bounds__(32) void k_res(const _Float16* __restrict__ xh,
    const _Float16* __restrict__ wp, const float* __restrict__ rb,
    float* __restrict__ res)
{
  const int lane = threadIdx.x;
  const int am  = lane & 15;
  const int hi  = (lane >> 4) & 1;
  const int kbA = hi ? 8 : 0;
  const int m   = blockIdx.x * 16 + am;
  const int bb  = m >> 12;
  const int l   = m & (SEQL - 1);
  const _Float16* arow = xh + ((size_t)bb * SEQL + l) * DIN;
  const v16h* bp = (const v16h*)wp;

  v16h a0 = cat16(*(const v8h*)(arow + kbA),      *(const v8h*)(arow + 16 + kbA));
  v16h a1 = cat16(*(const v8h*)(arow + 32 + kbA), *(const v8h*)(arow + 48 + kbA));

  for (int nt = 0; nt < 4; ++nt) {
    int n = nt * 16 + am;
    v16h b0 = bp[(nt * 2 + 0) * 32 + lane];
    v16h b1 = bp[(nt * 2 + 1) * 32 + lane];
    v8f acc = {};
    acc = wmma32(a0, b0, acc);
    acc = wmma32(a1, b1, acc);
    float bias = rb[n];
#pragma unroll
    for (int r = 0; r < 8; ++r) {
      int mm = blockIdx.x * 16 + r + (hi ? 8 : 0);
      int bb2 = mm >> 12, l2 = mm & (SEQL - 1);
      res[((size_t)bb2 * DIN + n) * SEQL + l2] = acc[r] + bias;
    }
  }
}

// ---------------- causal depthwise conv (k=4) + SiLU (f32 + f16 shadow) -----
__global__ __launch_bounds__(256) void k_dwconv(const float* __restrict__ xi,
    const float* __restrict__ cw, const float* __restrict__ cb,
    float* __restrict__ xc, _Float16* __restrict__ xch)
{
  int idx = blockIdx.x * blockDim.x + threadIdx.x;
  if (idx >= 4 * SEQL * DI) return;
  int d  = idx & (DI - 1);
  int bl = idx >> 7;
  int l  = bl & (SEQL - 1);
  int bb = bl >> 12;
  float s = cb[d];
#pragma unroll
  for (int k = 0; k < 4; ++k) {
    int ls = l - 3 + k;
    if (ls >= 0) s += xi[((size_t)(bb << 12) + ls) * DI + d] * cw[d * 4 + k];
  }
  float v = s / (1.f + __expf(-s));
  xc[idx]  = v;
  xch[idx] = (_Float16)v;
}

// ---------------- x_proj as WMMA: N=36 padded to 48, K=128 ------------------
__global__ __launch_bounds__(32) void k_xprojw(const _Float16* __restrict__ xch,
    const _Float16* __restrict__ wp, float* __restrict__ dt4,
    float* __restrict__ Bm, float* __restrict__ Cm)
{
  const int lane = threadIdx.x;
  const int am  = lane & 15;
  const int hi  = (lane >> 4) & 1;
  const int kbA = hi ? 8 : 0;
  const int m   = blockIdx.x * 16 + am;           // bl row directly
  const _Float16* arow = xch + (size_t)m * DI;
  const v16h* bp = (const v16h*)wp;

  v16h A[4];
#pragma unroll
  for (int kk = 0; kk < 4; ++kk)
    A[kk] = cat16(*(const v8h*)(arow + kk * 32 + kbA),
                  *(const v8h*)(arow + kk * 32 + 16 + kbA));

  for (int nt = 0; nt < 3; ++nt) {
    int n = nt * 16 + am;
    v8f acc = {};
#pragma unroll
    for (int kk = 0; kk < 4; ++kk)
      acc = wmma32(A[kk], bp[(nt * 4 + kk) * 32 + lane], acc);
    float* dst = nullptr; int col = 0, stride = 0;
    if (n < 4)       { dst = dt4; col = n;      stride = 4;  }
    else if (n < 20) { dst = Bm;  col = n - 4;  stride = 16; }
    else if (n < 36) { dst = Cm;  col = n - 20; stride = 16; }
    if (dst) {
#pragma unroll
      for (int r = 0; r < 8; ++r) {
        int mm = blockIdx.x * 16 + r + (hi ? 8 : 0);
        dst[(size_t)mm * stride + col] = acc[r];
      }
    }
  }
}

// ---------------- dt broadcast: softplus(dt4 @ dt_w^T + dt_b) ---------------
__global__ __launch_bounds__(256) void k_dt(const float* __restrict__ dt4,
    const float* __restrict__ dw, const float* __restrict__ db,
    float* __restrict__ dtf)
{
  int idx = blockIdx.x * blockDim.x + threadIdx.x;
  if (idx >= 4 * SEQL * DI) return;
  int d  = idx & (DI - 1);
  int bl = idx >> 7;
  float s = db[d];
#pragma unroll
  for (int r = 0; r < 4; ++r) s += dt4[bl * 4 + r] * dw[d * 4 + r];
  dtf[idx] = (s > 20.f) ? s : log1pf(__expf(s));
}

// ============ chunked selective scan (3 phases) ============
__global__ __launch_bounds__(256) void k_scan1(const float* __restrict__ xc,
    const float* __restrict__ dtf, const float* __restrict__ Bm,
    const float* __restrict__ A_log, float* __restrict__ S,
    float* __restrict__ P)
{
  int tid = blockIdx.x * blockDim.x + threadIdx.x;
  int s   = tid & 15;
  int gg  = tid >> 4;
  int c   = gg & (NCH - 1);
  int bd  = gg >> 6;
  int d   = bd & (DI - 1);
  int bb  = bd >> 7;
  float a = -__expf(A_log[d * NST + s]);
  float h = 0.f, dts = 0.f;
  size_t bl0 = (size_t)bb * SEQL + c * CL;
  for (int t = 0; t < CL; ++t) {
    size_t bl = bl0 + t;
    float dt = dtf[bl * DI + d];
    float xv = xc [bl * DI + d];
    float Bv = Bm [bl * NST + s];
    h = __expf(dt * a) * h + dt * Bv * xv;
    dts += dt;
  }
  S[(size_t)gg * NST + s] = h;
  P[(size_t)gg * NST + s] = __expf(a * dts);
}

__global__ __launch_bounds__(256) void k_scan2(const float* __restrict__ S,
    const float* __restrict__ P, float* __restrict__ H0)
{
  int tid = blockIdx.x * blockDim.x + threadIdx.x;
  int s   = tid & 15;
  int bd  = tid >> 4;
  float h = 0.f;
  for (int c = 0; c < NCH; ++c) {
    size_t idx = ((size_t)((bd << 6) | c)) * NST + s;
    H0[idx] = h;
    h = P[idx] * h + S[idx];
  }
}

__global__ __launch_bounds__(256) void k_scan3(const float* __restrict__ xc,
    const float* __restrict__ dtf, const float* __restrict__ Bm,
    const float* __restrict__ Cm, const float* __restrict__ zb,
    const float* __restrict__ A_log, const float* __restrict__ Dp,
    const float* __restrict__ H0, _Float16* __restrict__ yh)
{
  int tid = blockIdx.x * blockDim.x + threadIdx.x;
  int s   = tid & 15;
  int gg  = tid >> 4;
  int c   = gg & (NCH - 1);
  int bd  = gg >> 6;
  int d   = bd & (DI - 1);
  int bb  = bd >> 7;
  float a  = -__expf(A_log[d * NST + s]);
  float Dv = Dp[d];
  float h  = H0[(size_t)gg * NST + s];
  size_t bl0 = (size_t)bb * SEQL + c * CL;
  for (int t = 0; t < CL; ++t) {
    size_t bl = bl0 + t;
    if (t + 8 < CL) {   // global_prefetch_b8 of upcoming rows
      __builtin_prefetch(&dtf[(bl + 8) * DI + d], 0, 1);
      __builtin_prefetch(&xc [(bl + 8) * DI + d], 0, 1);
    }
    float dt = dtf[bl * DI + d];
    float xv = xc [bl * DI + d];
    float Bv = Bm [bl * NST + s];
    float Cv = Cm [bl * NST + s];
    h = __expf(dt * a) * h + dt * Bv * xv;
    float acc = Cv * h;
#pragma unroll
    for (int off = 8; off; off >>= 1) acc += __shfl_xor(acc, off, 32);
    if (s == 0) {
      float zv = zb[bl * DI + d];
      yh[bl * DI + d] = (_Float16)((acc + xv * Dv) * (zv / (1.f + __expf(-zv))));
    }
  }
}

// ---------------- out_proj + concat, ycat stored [b][l][ch] f16 -------------
__global__ __launch_bounds__(32) void k_outproj(const _Float16* __restrict__ yh,
    const _Float16* __restrict__ wp, _Float16* __restrict__ ycat, int dir)
{
  const int lane = threadIdx.x;
  const int am  = lane & 15;
  const int hi  = (lane >> 4) & 1;
  const int kbA = hi ? 8 : 0;
  const int m   = blockIdx.x * 16 + am;
  const int bb  = m >> 12;
  const int l   = m & (SEQL - 1);
  const _Float16* arow = yh + ((size_t)bb * SEQL + l) * DI;
  const v16h* bp = (const v16h*)wp;

  v16h A[4];
#pragma unroll
  for (int kk = 0; kk < 4; ++kk)
    A[kk] = cat16(*(const v8h*)(arow + kk * 32 + kbA),
                  *(const v8h*)(arow + kk * 32 + 16 + kbA));

  for (int nt = 0; nt < 4; ++nt) {
    int n = nt * 16 + am;
    v8f acc = {};
#pragma unroll
    for (int kk = 0; kk < 4; ++kk)
      acc = wmma32(A[kk], bp[(nt * 4 + kk) * 32 + lane], acc);
#pragma unroll
    for (int r = 0; r < 8; ++r) {
      int mm = blockIdx.x * 16 + r + (hi ? 8 : 0);
      int bb2 = mm >> 12, l2 = mm & (SEQL - 1);
      ycat[((size_t)bb2 * SEQL + l2) * DI + dir * 64 + n] = (_Float16)acc[r];
    }
  }
}

// ---------------- 3x3 conv: implicit GEMM, vectorized f16 A loads -----------
__global__ __launch_bounds__(32) void k_conv3(const _Float16* __restrict__ ycat,
    const _Float16* __restrict__ wp, const float* __restrict__ cb,
    float* __restrict__ co)
{
  const int lane = threadIdx.x;
  const int am  = lane & 15;
  const int hi  = (lane >> 4) & 1;
  const int kbA = hi ? 8 : 0;
  const int m   = blockIdx.x * 16 + am;
  const int bb  = m >> 12;
  const int l   = m & (SEQL - 1);
  const int hh  = l >> 6;
  const int wwp = l & 63;
  const v16h* cp = (const v16h*)wp;
  const _Float16* base = ycat + ((size_t)bb * SEQL) * DI;

  v8f acc[4] = {{}, {}, {}, {}};
  for (int ky = 0; ky < 3; ++ky) {
    int h2 = hh + ky - 1;
    bool okh = (unsigned)h2 < 64u;
    for (int kx = 0; kx < 3; ++kx) {
      int w2 = wwp + kx - 1;
      bool ok = okh && ((unsigned)w2 < 64u);
      int pos = ky * 3 + kx;
      const _Float16* arow = base + (size_t)(h2 * 64 + w2) * DI;
      for (int kk = 0; kk < 4; ++kk) {
        v16h a = {};
        if (ok)
          a = cat16(*(const v8h*)(arow + kk * 32 + kbA),
                    *(const v8h*)(arow + kk * 32 + 16 + kbA));
#pragma unroll
        for (int nt = 0; nt < 4; ++nt)
          acc[nt] = wmma32(a, cp[((pos * 4 + kk) * 4 + nt) * 32 + lane], acc[nt]);
      }
    }
  }
#pragma unroll
  for (int nt = 0; nt < 4; ++nt) {
    int n = nt * 16 + am;
    float bias = cb[n];
#pragma unroll
    for (int r = 0; r < 8; ++r) {
      int mm = blockIdx.x * 16 + r + (hi ? 8 : 0);
      int bb2 = mm >> 12, l2 = mm & (SEQL - 1);
      co[((size_t)bb2 * 64 + n) * SEQL + l2] = acc[nt][r] + bias;
    }
  }
}

// ---------------- batchnorm statistics (per out channel) --------------------
__global__ __launch_bounds__(256) void k_bnstats(const float* __restrict__ co,
                                                 float* __restrict__ stats)
{
  __shared__ float sh[512];
  int c = blockIdx.x;
  float s = 0.f, s2 = 0.f;
  for (int i = threadIdx.x; i < 4 * SEQL; i += 256) {
    int bb = i >> 12, l = i & (SEQL - 1);
    float v = co[((size_t)bb * 64 + c) * SEQL + l];
    s += v; s2 += v * v;
  }
  sh[threadIdx.x] = s; sh[256 + threadIdx.x] = s2;
  __syncthreads();
  for (int off = 128; off; off >>= 1) {
    if (threadIdx.x < off) {
      sh[threadIdx.x]       += sh[threadIdx.x + off];
      sh[256 + threadIdx.x] += sh[256 + threadIdx.x + off];
    }
    __syncthreads();
  }
  if (threadIdx.x == 0) {
    float mean = sh[0]   * (1.f / 16384.f);
    float var  = sh[256] * (1.f / 16384.f) - mean * mean;
    stats[c] = mean; stats[64 + c] = var;
  }
}

// ---------------- BN apply + residual + leaky ReLU --------------------------
__global__ __launch_bounds__(1024) void k_final(const float* __restrict__ co,
    const float* __restrict__ res, const float* __restrict__ stats,
    const float* __restrict__ g, const float* __restrict__ be,
    float* __restrict__ out)
{
  int idx = blockIdx.x * blockDim.x + threadIdx.x;
  if (idx >= 4 * 64 * SEQL) return;
  int c = (idx >> 12) & 63;
  float v = (co[idx] - stats[c]) * rsqrtf(stats[64 + c] + 1e-5f) * g[c]
            + be[c] + res[idx];
  out[idx] = (v >= 0.f) ? v : 0.01f * v;
}

extern "C" void kernel_launch(void* const* d_in, const int* in_sizes, int n_in,
                              void* d_out, int out_size, void* d_ws, size_t ws_size,
                              hipStream_t stream)
{
  (void)in_sizes; (void)n_in; (void)out_size; (void)ws_size;
  const float* x      = (const float*)d_in[0];
  const float* conv_w = (const float*)d_in[19];
  const float* conv_b = (const float*)d_in[20];
  const float* bn_g   = (const float*)d_in[21];
  const float* bn_b   = (const float*)d_in[22];
  const float* res_w  = (const float*)d_in[23];
  const float* res_b  = (const float*)d_in[24];

  // f32 region
  float* ws    = (float*)d_ws;
  float* res   = ws;                       // 1,048,576
  float* co    = res  + 1048576;           // 1,048,576
  float* stats = co   + 1048576;           // 256
  float* xi    = stats + 256;              // 2,097,152
  float* zbuf  = xi    + 2097152;          // 2,097,152
  float* xc    = zbuf  + 2097152;          // 2,097,152
  float* dtf   = xc    + 2097152;          // 2,097,152
  float* dt4   = dtf   + 2097152;          // 65,536
  float* Bm    = dt4   + 65536;            // 262,144
  float* Cm    = Bm    + 262144;           // 262,144
  float* Sb    = Cm    + 262144;           // 524,288
  float* Pb    = Sb    + 524288;           // 524,288
  float* H0b   = Pb    + 524288;           // 524,288
  // f16 region (starts 256B-aligned)
  _Float16* h16   = (_Float16*)(H0b + 524288);
  _Float16* ycat  = h16;                   // 2,097,152 halves (b,L,128)
  _Float16* yh    = ycat + 2097152;        // 2,097,152
  _Float16* xch   = yh   + 2097152;        // 2,097,152
  _Float16* xh    = xch  + 2097152;        // 1,048,576  (b,L,64)
  _Float16* inpk  = xh   + 1048576;        // 16*2*32*16  = 16384
  _Float16* xpk   = inpk + 16384;          // 3*4*32*16   = 6144
  _Float16* owpk  = xpk  + 6144;           // 4*4*32*16   = 8192
  _Float16* respk = owpk + 8192;           // 4*2*32*16   = 4096
  _Float16* cvpk  = respk + 4096;          // 9*4*4*32*16 = 73728

  k_packx<<<4096, 256, 0, stream>>>(x, xh);
  k_pack<<<1, 256, 0, stream>>>(res_w, respk, 4, 2, 64, 63);
  k_packconv<<<18, 256, 0, stream>>>(conv_w, cvpk);
  k_res<<<1024, 32, 0, stream>>>(xh, respk, res_b, res);

  for (int dir = 0; dir < 2; ++dir) {
    const float* in_w  = (const float*)d_in[1 + dir * 9];
    const float* dcw   = (const float*)d_in[2 + dir * 9];
    const float* dcb   = (const float*)d_in[3 + dir * 9];
    const float* pw    = (const float*)d_in[4 + dir * 9];
    const float* dw    = (const float*)d_in[5 + dir * 9];
    const float* db    = (const float*)d_in[6 + dir * 9];
    const float* A_log = (const float*)d_in[7 + dir * 9];
    const float* Dp    = (const float*)d_in[8 + dir * 9];
    const float* ow    = (const float*)d_in[9 + dir * 9];

    k_pack<<<4, 256, 0, stream>>>(in_w, inpk, 16, 2, 64, 255);
    k_pack<<<2, 256, 0, stream>>>(pw,   xpk,   3, 4, 128, 35);
    k_pack<<<2, 256, 0, stream>>>(ow,   owpk,  4, 4, 128, 63);

    k_inproj<<<1024, 32, 0, stream>>>(xh, inpk, xi, zbuf, dir);
    k_dwconv<<<8192, 256, 0, stream>>>(xi, dcw, dcb, xc, xch);
    k_xprojw<<<1024, 32, 0, stream>>>(xch, xpk, dt4, Bm, Cm);
    k_dt    <<<8192, 256, 0, stream>>>(dt4, dw, db, dtf);
    k_scan1 <<<2048, 256, 0, stream>>>(xc, dtf, Bm, A_log, Sb, Pb);
    k_scan2 <<<32, 256, 0, stream>>>(Sb, Pb, H0b);
    k_scan3 <<<2048, 256, 0, stream>>>(xc, dtf, Bm, Cm, zbuf, A_log, Dp, H0b, yh);
    k_outproj<<<1024, 32, 0, stream>>>(yh, owpk, ycat, dir);
  }

  k_conv3  <<<1024, 32, 0, stream>>>(ycat, cvpk, conv_b, co);
  k_bnstats<<<64, 256, 0, stream>>>(co, stats);
  k_final  <<<1024, 1024, 0, stream>>>(co, res, stats, bn_g, bn_b, (float*)d_out);
}
